// ActionDecoder_21053929684997
// MI455X (gfx1250) — matrix operand order repaired and checked by reading, not compiled, not run on hardware
//
#include <hip/hip_runtime.h>
#include <hip/hip_bf16.h>
#include <math.h>

#define BATCH 4096
#define EMBED 8192
#define U0    1024
#define U1    256
#define EGO   3
#define NHEAD 6
#define W1ROWS (U0 + EGO)   // 1027

typedef __bf16 v16bf __attribute__((ext_vector_type(16)));
typedef float  v8f   __attribute__((ext_vector_type(8)));

union FragBf { v16bf v; unsigned int u[8]; };

__device__ __forceinline__ void bf16_split(float f, __bf16& hi, __bf16& lo) {
  hi = (__bf16)f;
  lo = (__bf16)(f - (float)hi);
}

__device__ __forceinline__ v8f wmma_bf16(v16bf a, v16bf b, v8f c) {
  return __builtin_amdgcn_wmma_f32_16x16x32_bf16(false, a, false, b,
                                                 (short)0, c, false, false);
}

// ---------------------------------------------------------------------------
// Kernel 1: h = relu(x @ W_fc + b_fc)   [4096 x 8192] * [8192 x 1024]
// 128x128 tile per 256-thread block, bf16x3 split-precision WMMA, f32 accum.
// ---------------------------------------------------------------------------
__global__ __launch_bounds__(256) void trunk_fc_kernel(
    const float* __restrict__ x, const float* __restrict__ Wfc,
    const float* __restrict__ bfc, float* __restrict__ h)
{
  __shared__ __bf16 Ahi[128][40];
  __shared__ __bf16 Alo[128][40];
  __shared__ __bf16 Bhi[128][40];   // transposed: [n][k]
  __shared__ __bf16 Blo[128][40];

  const int t    = threadIdx.x;
  const int lane = t & 31;
  const int wave = t >> 5;
  const int hf   = lane >> 4;       // 0 or 1 (which lane half)
  const int ln16 = lane & 15;
  const int mBase = blockIdx.x * 128;
  const int nBase = blockIdx.y * 128;
  const int wm = wave >> 1;         // 0..3 -> 32-row slice
  const int wn = wave & 1;          // 0..1 -> 64-col slice

  v8f acc[2][4] = {};

  // loader coordinates
  const int aRow = t >> 1;              // 0..127
  const int aCol = (t & 1) << 4;        // 0 or 16
  const int bRow = t >> 3;              // 0..31 (k)
  const int bCol = (t & 7) << 4;        // 0..112 (n)
  const float* aG = x   + (size_t)(mBase + aRow) * EMBED + aCol;
  const float* bG = Wfc + (size_t)bRow * U0 + (nBase + bCol);

  for (int kk = 0; kk < EMBED; kk += 32) {
    // ---- stage A tile (128x32 f32 -> hi/lo bf16) ----
    const float4* a4 = (const float4*)(aG + kk);
#pragma unroll
    for (int q = 0; q < 4; ++q) {
      float4 f = a4[q];
      float fv[4] = {f.x, f.y, f.z, f.w};
#pragma unroll
      for (int j = 0; j < 4; ++j) {
        __bf16 hi, lo; bf16_split(fv[j], hi, lo);
        Ahi[aRow][aCol + q * 4 + j] = hi;
        Alo[aRow][aCol + q * 4 + j] = lo;
      }
    }
    // ---- stage B tile (32x128 f32 -> transposed hi/lo bf16) ----
    const float4* b4 = (const float4*)(bG + (size_t)kk * U0);
#pragma unroll
    for (int q = 0; q < 4; ++q) {
      float4 f = b4[q];
      float fv[4] = {f.x, f.y, f.z, f.w};
#pragma unroll
      for (int j = 0; j < 4; ++j) {
        __bf16 hi, lo; bf16_split(fv[j], hi, lo);
        Bhi[bCol + q * 4 + j][bRow] = hi;
        Blo[bCol + q * 4 + j][bRow] = lo;
      }
    }
    if (kk + 32 < EMBED) {
      __builtin_prefetch(aG + kk + 32, 0, 0);                 // global_prefetch_b8
      __builtin_prefetch(bG + (size_t)(kk + 32) * U0, 0, 0);
    }
    __syncthreads();

    // ---- build A fragments (contiguous k order: lanes 0-15 K0-15, lanes 16-31 K16-31) ----
    FragBf ahi[2], alo[2];
#pragma unroll
    for (int mt = 0; mt < 2; ++mt) {
      int r = wm * 32 + mt * 16 + ln16;
#pragma unroll
      for (int j = 0; j < 8; ++j) {
        int k0 = 16 * hf + 2 * j;
        ahi[mt].u[j] = *(const unsigned int*)&Ahi[r][k0];
        alo[mt].u[j] = *(const unsigned int*)&Alo[r][k0];
      }
    }
    // ---- B fragments + 3-term WMMA accumulation ----
#pragma unroll
    for (int nt = 0; nt < 4; ++nt) {
      FragBf bhi, blo;
      int c = wn * 64 + nt * 16 + ln16;
#pragma unroll
      for (int j = 0; j < 8; ++j) {
        int k0 = 16 * hf + 2 * j;
        bhi.u[j] = *(const unsigned int*)&Bhi[c][k0];
        blo.u[j] = *(const unsigned int*)&Blo[c][k0];
      }
#pragma unroll
      for (int mt = 0; mt < 2; ++mt) {
        acc[mt][nt] = wmma_bf16(ahi[mt].v, bhi.v, acc[mt][nt]);
        acc[mt][nt] = wmma_bf16(ahi[mt].v, blo.v, acc[mt][nt]);
        acc[mt][nt] = wmma_bf16(alo[mt].v, bhi.v, acc[mt][nt]);
      }
    }
    __syncthreads();
  }

  // ---- epilogue: bias + relu, store h ----
#pragma unroll
  for (int nt = 0; nt < 4; ++nt) {
    int n = nBase + wn * 64 + nt * 16 + ln16;
    float bias = bfc[n];
#pragma unroll
    for (int mt = 0; mt < 2; ++mt) {
#pragma unroll
      for (int e = 0; e < 8; ++e) {
        int m = mBase + wm * 32 + mt * 16 + e + 8 * hf;
        float val = acc[mt][nt][e] + bias;
        h[(size_t)m * U0 + n] = val > 0.f ? val : 0.f;
      }
    }
  }
}

// ---------------------------------------------------------------------------
// Kernel 2: per head: z1 = relu([h,ego] @ W1[head] + b1[head]);
//           z2 = z1 @ W2[head] + b2[head]; select head == command-1;
//           mean/std transform; write out.
// 32 samples x 256 cols per block; grid = (128, 6 heads).
// ---------------------------------------------------------------------------
__global__ __launch_bounds__(256) void heads_kernel(
    const float* __restrict__ h, const float* __restrict__ ego,
    const int* __restrict__ command, const float* __restrict__ W1,
    const float* __restrict__ b1, const float* __restrict__ W2,
    const float* __restrict__ b2, float* __restrict__ out)
{
  __shared__ __bf16 Ahi[32][40];
  __shared__ __bf16 Alo[32][40];
  union ShB {
    struct { __bf16 hi[256][40]; __bf16 lo[256][40]; } w;  // transposed W1 tile
    float z1[32][260];                                     // reused after GEMM
  };
  __shared__ ShB shb;
  __shared__ float w2s[U1 * 4];

  const int t    = threadIdx.x;
  const int lane = t & 31;
  const int wave = t >> 5;
  const int hf   = lane >> 4;
  const int ln16 = lane & 15;
  const int mBase = blockIdx.x * 32;
  const int head  = blockIdx.y;
  const int mOff = (wave >> 2) * 16;   // 0 or 16
  const int nOff = (wave & 3) * 64;    // 0,64,128,192

  for (int i = t; i < U1 * 4; i += 256)
    w2s[i] = W2[(size_t)head * U1 * 4 + i];

  v8f acc[4] = {};

  const int aRow = t >> 3;             // 0..31
  const int aCol = (t & 7) << 2;       // 0..28
  const int bRow = t >> 3;             // 0..31 (k)
  const int bCol = (t & 7) << 5;       // 0..224 (n)
  const float* aG = h  + (size_t)(mBase + aRow) * U0 + aCol;
  const float* bG = W1 + ((size_t)head * W1ROWS + bRow) * U1 + bCol;

  for (int kk = 0; kk < U0; kk += 32) {
    float4 fa = *(const float4*)(aG + kk);
    {
      float fv[4] = {fa.x, fa.y, fa.z, fa.w};
#pragma unroll
      for (int j = 0; j < 4; ++j) {
        __bf16 hi, lo; bf16_split(fv[j], hi, lo);
        Ahi[aRow][aCol + j] = hi;
        Alo[aRow][aCol + j] = lo;
      }
    }
    const float4* b4 = (const float4*)(bG + (size_t)kk * U1);
#pragma unroll
    for (int q = 0; q < 8; ++q) {
      float4 f = b4[q];
      float fv[4] = {f.x, f.y, f.z, f.w};
#pragma unroll
      for (int j = 0; j < 4; ++j) {
        __bf16 hi, lo; bf16_split(fv[j], hi, lo);
        shb.w.hi[bCol + q * 4 + j][bRow] = hi;
        shb.w.lo[bCol + q * 4 + j][bRow] = lo;
      }
    }
    __syncthreads();

    FragBf ahi, alo;
    int r = mOff + ln16;
#pragma unroll
    for (int j = 0; j < 8; ++j) {
      int k0 = 16 * hf + 2 * j;
      ahi.u[j] = *(const unsigned int*)&Ahi[r][k0];
      alo.u[j] = *(const unsigned int*)&Alo[r][k0];
    }
#pragma unroll
    for (int nt = 0; nt < 4; ++nt) {
      FragBf bhi, blo;
      int c = nOff + nt * 16 + ln16;
#pragma unroll
      for (int j = 0; j < 8; ++j) {
        int k0 = 16 * hf + 2 * j;
        bhi.u[j] = *(const unsigned int*)&shb.w.hi[c][k0];
        blo.u[j] = *(const unsigned int*)&shb.w.lo[c][k0];
      }
      acc[nt] = wmma_bf16(ahi.v, bhi.v, acc[nt]);
      acc[nt] = wmma_bf16(ahi.v, blo.v, acc[nt]);
      acc[nt] = wmma_bf16(alo.v, bhi.v, acc[nt]);
    }
    __syncthreads();
  }

  // ---- epilogue: ego rows (K = 1024..1026) + bias + relu -> z1 in LDS ----
#pragma unroll
  for (int nt = 0; nt < 4; ++nt) {
    int n = nOff + nt * 16 + ln16;
    float bias = b1[(size_t)head * U1 + n];
    float w1e0 = W1[((size_t)head * W1ROWS + U0 + 0) * U1 + n];
    float w1e1 = W1[((size_t)head * W1ROWS + U0 + 1) * U1 + n];
    float w1e2 = W1[((size_t)head * W1ROWS + U0 + 2) * U1 + n];
#pragma unroll
    for (int e = 0; e < 8; ++e) {
      int r = mOff + e + 8 * hf;
      int s = mBase + r;
      float val = acc[nt][e] + bias
                + ego[s * 3 + 0] * w1e0
                + ego[s * 3 + 1] * w1e1
                + ego[s * 3 + 2] * w1e2;
      shb.z1[r][n] = val > 0.f ? val : 0.f;
    }
  }
  __syncthreads();

  // ---- stage 2: z2 = z1 @ W2 + b2; select + transform + store ----
  if (t < 128) {
    int s = t >> 2, d = t & 3;
    float a2 = b2[head * 4 + d];
#pragma unroll 8
    for (int k = 0; k < U1; ++k)
      a2 += shb.z1[s][k] * w2s[k * 4 + d];
    int sg = mBase + s;
    if (command[sg] - 1 == head) {
      float o;
      if (d < 2) {
        o = 5.0f * tanhf(a2 * 0.2f);                  // MEAN_SCALE * tanh(m/5)
      } else {
        float xs = a2 + 4.9932389f;                   // + log(exp(5)-1)
        o = (xs > 20.0f ? xs : log1pf(expf(xs))) + 1e-4f;
      }
      out[(size_t)sg * 4 + d] = o;
    }
  }
}

// ---------------------------------------------------------------------------
extern "C" void kernel_launch(void* const* d_in, const int* in_sizes, int n_in,
                              void* d_out, int out_size, void* d_ws, size_t ws_size,
                              hipStream_t stream) {
  (void)in_sizes; (void)n_in; (void)out_size; (void)ws_size;
  const float* x   = (const float*)d_in[0];
  const int*   cmd = (const int*)d_in[1];
  const float* eg  = (const float*)d_in[2];
  const float* Wfc = (const float*)d_in[3];
  const float* bfc = (const float*)d_in[4];
  const float* W1  = (const float*)d_in[5];
  const float* b1  = (const float*)d_in[6];
  const float* W2  = (const float*)d_in[7];
  const float* b2  = (const float*)d_in[8];
  float* out = (float*)d_out;
  float* hbuf = (float*)d_ws;     // 4096*1024 f32 = 16 MB scratch

  dim3 gA(BATCH / 128, U0 / 128);           // 32 x 8
  trunk_fc_kernel<<<gA, 256, 0, stream>>>(x, Wfc, bfc, hbuf);

  dim3 gB(BATCH / 32, NHEAD);               // 128 x 6
  heads_kernel<<<gB, 256, 0, stream>>>(hbuf, eg, cmd, W1, b1, W2, b2, out);
}
